// Decoder_49452253446715
// MI455X (gfx1250) — compile-verified, matrix operand order checked by reading, not run on hardware
//
#include <hip/hip_runtime.h>
#include <hip/hip_bf16.h>

typedef __attribute__((ext_vector_type(16))) _Float16 v16h;
typedef __attribute__((ext_vector_type(8)))  _Float16 v8h;
typedef __attribute__((ext_vector_type(8)))  float    v8f;

#define B_   2048
#define CH_  64
#define T_   168
#define H_   128
#define A_   64
#define P_   24
#define KG_  320   // padded gates K: 192 (ctx|tp|pad) + 128 (h)
#define KI_  192   // padded inp K
#define KX_  192   // padded x-attention K

__device__ __forceinline__ float sigm(float v) { return 1.0f / (1.0f + __expf(-v)); }

// Load one 16-half WMMA fragment (A or B operand) given the CDNA5 16-bit
// layout: halves j=0..7 at k0, halves j=8..15 at k0+16 (k0 already includes
// the lane>=16 offset of +8). Two 16-byte loads.
__device__ __forceinline__ v16h load_frag(const _Float16* p) {
    v16h a;
    ((v8h*)&a)[0] = *(const v8h*)(p);
    ((v8h*)&a)[1] = *(const v8h*)(p + 16);
    return a;
}

// ---------------------------------------------------------------------------
// Pre-pass 1: pack weights to f16 in WMMA-friendly (N-major, K-contiguous,
// zero-padded) layouts; combine biases.
// ---------------------------------------------------------------------------
__global__ void pack_weights(const float* __restrict__ Wa1,
                             const float* __restrict__ Wih,
                             const float* __restrict__ Whh,
                             const float* __restrict__ bih,
                             const float* __restrict__ bhh,
                             const float* __restrict__ Wf1,
                             _Float16* __restrict__ Wg,      // [512][320]
                             _Float16* __restrict__ Wa1hcT,  // [64][256]
                             _Float16* __restrict__ Wa1xT,   // [64][192]
                             _Float16* __restrict__ Wf1T,    // [64][128]
                             float*    __restrict__ bsum)    // [512]
{
    int tid = blockIdx.x * blockDim.x + threadIdx.x;
    int str = gridDim.x * blockDim.x;
    // gates weight: n in [0,512), k in [0,320): k<192 -> Wih[n][k] (k<169) / 0 ; else Whh[n][k-192]
    for (int i = tid; i < 512 * KG_; i += str) {
        int n = i / KG_, k = i - n * KG_;
        float v;
        if (k < KI_) v = (k < T_ + 1) ? Wih[n * (T_ + 1) + k] : 0.0f;
        else         v = Whh[n * H_ + (k - KI_)];
        Wg[i] = (_Float16)v;
    }
    // attention [h|c] weight, transposed: [a][k], k in [0,256)
    for (int i = tid; i < A_ * 2 * H_; i += str) {
        int a = i / (2 * H_), k = i - a * (2 * H_);
        Wa1hcT[i] = (_Float16)Wa1[k * A_ + a];
    }
    // attention x weight, transposed + padded: [a][k], k in [0,192)
    for (int i = tid; i < A_ * KX_; i += str) {
        int a = i / KX_, k = i - a * KX_;
        Wa1xT[i] = (k < T_) ? (_Float16)Wa1[(2 * H_ + k) * A_ + a] : (_Float16)0.0f;
    }
    // FF1 weight transposed: [n][k], k in [0,128)
    for (int i = tid; i < 64 * H_; i += str) {
        int n = i / H_, k = i - n * H_;
        Wf1T[i] = (_Float16)Wf1[k * 64 + n];
    }
    for (int i = tid; i < 512; i += str) bsum[i] = bih[i] + bhh[i];
}

// ---------------------------------------------------------------------------
// Pre-pass 2: xw[b*CH+ch][a] = sum_t x[b,ch,t] * Wa1[2H+t][a]
// One 131072x192x64 WMMA GEMM (step-invariant part of the attention).
// ---------------------------------------------------------------------------
__global__ __launch_bounds__(128) void xw_kernel(const float* __restrict__ x,
                                                 const _Float16* __restrict__ Wa1xT,
                                                 float* __restrict__ xw)
{
    const int wv = threadIdx.x >> 5;          // 4 waves: one N-tile of 16 each
    const int ln = threadIdx.x & 31;
    const int row0 = blockIdx.x * 16;         // 8192 blocks
    const int m = ln & 15;
    const int off = (ln & 16) ? 8 : 0;
    const long gr = (long)(row0 + m) * T_;

    v8f acc = {};
#pragma unroll
    for (int kc = 0; kc < KX_ / 32; ++kc) {
        const int k0 = kc * 32 + off;
        v16h a;
#pragma unroll
        for (int j = 0; j < 16; ++j) {
            int k = k0 + ((j & 8) ? 16 : 0) + (j & 7);
            float v = (k < T_) ? x[gr + k] : 0.0f;
            a[j] = (_Float16)v;
        }
        v16h b = load_frag(Wa1xT + (wv * 16 + m) * KX_ + k0);
        acc = __builtin_amdgcn_wmma_f32_16x16x32_f16(false, a, false, b,
                                                     (short)0, acc, false, false);
    }
#pragma unroll
    for (int r = 0; r < 8; ++r)
        xw[(long)(row0 + r + off) * A_ + wv * 16 + m] = acc[r];
}

// ---------------------------------------------------------------------------
// Main persistent decoder: 128 WGs x 256 threads, 16 batch rows per WG,
// 24 recurrent steps with LDS-resident state.
// ---------------------------------------------------------------------------
__global__ __launch_bounds__(256) void decoder_main(
    const float* __restrict__ x,    const float* __restrict__ his,
    const float* __restrict__ ba1,  const float* __restrict__ Wa2,
    const float* __restrict__ ba2p, const float* __restrict__ bf1,
    const float* __restrict__ Wf2,  const float* __restrict__ bf2p,
    const _Float16* __restrict__ Wg,     const _Float16* __restrict__ Wa1hcT,
    const _Float16* __restrict__ Wf1T,   const float* __restrict__ bsum,
    const float* __restrict__ xw,        float* __restrict__ out)
{
    __shared__ alignas(16) _Float16 hc[16][256];   // [h(128) | c(128)] f16 state
    __shared__ alignas(16) _Float16 inp[16][KI_];  // [ctx(168) | tp | 0-pad]
    __shared__ float cst[16][H_];                  // f32 cell carry
    __shared__ float ahc[16][A_];                  // [h|c]@Wa1hc
    __shared__ float sp[16][CH_];                  // scores -> probs
    __shared__ float rel[16][64];                  // relu(h@Wf1+bf1)
    __shared__ float tp[16];
    __shared__ float ba1_s[64], Wa2_s[64], bf1_s[64], Wf2_s[64], bsum_s[512];

    const int tid = threadIdx.x;
    const int wv  = tid >> 5;
    const int ln  = tid & 31;
    const int m   = ln & 15;
    const int off = (ln & 16) ? 8 : 0;
    const int b0  = blockIdx.x * 16;

    for (int i = tid; i < 16 * 256; i += 256) hc[i >> 8][i & 255] = (_Float16)0.0f;
    for (int i = tid; i < 16 * H_; i += 256)  cst[i >> 7][i & 127] = 0.0f;
    if (tid < 64) { ba1_s[tid] = ba1[tid]; Wa2_s[tid] = Wa2[tid];
                    bf1_s[tid] = bf1[tid]; Wf2_s[tid] = Wf2[tid]; }
    for (int i = tid; i < 512; i += 256) bsum_s[i] = bsum[i];
    if (tid < 16) tp[tid] = his[(long)(b0 + tid) * T_ + (T_ - 1)];
    const float ba2v = ba2p[0], bf2v = bf2p[0];
    __syncthreads();

    for (int p = 0; p < P_; ++p) {
        // ---- Phase A: ahc = [h|c](16x256) @ Wa1hc(256x64), waves 0..3 ----
        if (wv < 4) {
            v8f acc = {};
#pragma unroll
            for (int kc = 0; kc < 8; ++kc) {
                v16h a = load_frag(&hc[m][kc * 32 + off]);
                v16h b = load_frag(Wa1hcT + (wv * 16 + m) * 256 + kc * 32 + off);
                acc = __builtin_amdgcn_wmma_f32_16x16x32_f16(false, a, false, b,
                                                             (short)0, acc, false, false);
            }
#pragma unroll
            for (int r = 0; r < 8; ++r) ahc[r + off][wv * 16 + m] = acc[r];
        }
        __syncthreads();

        // ---- Phase B1: attention scores (VALU, transcendental) ----
        for (int pr = tid; pr < 16 * CH_; pr += 256) {
            int b = pr >> 6, ch = pr & 63;
            const float* xwp = xw + ((long)(b0 + b) * CH_ + ch) * A_;
            float acc = ba2v;
#pragma unroll 8
            for (int a = 0; a < A_; ++a)
                acc += Wa2_s[a] * tanhf(xwp[a] + ahc[b][a] + ba1_s[a]);
            sp[b][ch] = acc;
        }
        __syncthreads();

        // ---- Phase B2: softmax over CH=64 (2 rows per wave) ----
        {
#pragma unroll
            for (int bb = wv * 2; bb < wv * 2 + 2; ++bb) {
                float v0 = sp[bb][ln], v1 = sp[bb][ln + 32];
                float mx = fmaxf(v0, v1);
#pragma unroll
                for (int o = 16; o; o >>= 1) mx = fmaxf(mx, __shfl_xor(mx, o, 32));
                float e0 = __expf(v0 - mx), e1 = __expf(v1 - mx);
                float sm = e0 + e1;
#pragma unroll
                for (int o = 16; o; o >>= 1) sm += __shfl_xor(sm, o, 32);
                float inv = 1.0f / sm;
                sp[bb][ln] = e0 * inv; sp[bb][ln + 32] = e1 * inv;
            }
        }
        __syncthreads();

        // ---- Phase B3: ctx = p @ x[b]  (x is L2-resident: 88MB < 192MB) ----
        for (int i = tid; i < 16 * T_; i += 256) {
            int b = i / T_, t = i - b * T_;
            const float* xp = x + (long)(b0 + b) * CH_ * T_ + t;
            float acc = 0.0f;
#pragma unroll 8
            for (int ch = 0; ch < CH_; ++ch) acc += sp[b][ch] * xp[ch * T_];
            inp[b][t] = (_Float16)acc;
        }
        for (int i = tid; i < 16 * (KI_ - T_); i += 256) {
            int b = i / (KI_ - T_), t = T_ + i % (KI_ - T_);
            inp[b][t] = (t == T_) ? (_Float16)tp[b] : (_Float16)0.0f;
        }
        __syncthreads();

        // ---- Phase C: gates(16x512) = [inp|h](16x320) @ Wg^T, all 8 waves.
        // Wave w owns n = w*16..w*16+15 of ALL four gates -> LSTM in registers.
        v8f g0 = {}, g1 = {}, g2 = {}, g3 = {};
        const int n = wv * 16 + m;
#pragma unroll
        for (int kc = 0; kc < KG_ / 32; ++kc) {
            const int k0 = kc * 32 + off;
            const _Float16* ap = (kc < 6) ? &inp[m][k0] : &hc[m][k0 - KI_];
            v16h a = load_frag(ap);
            v16h w0 = load_frag(Wg + (long)(0 * 128 + n) * KG_ + k0);
            v16h w1 = load_frag(Wg + (long)(1 * 128 + n) * KG_ + k0);
            v16h w2 = load_frag(Wg + (long)(2 * 128 + n) * KG_ + k0);
            v16h w3 = load_frag(Wg + (long)(3 * 128 + n) * KG_ + k0);
            g0 = __builtin_amdgcn_wmma_f32_16x16x32_f16(false, a, false, w0, (short)0, g0, false, false);
            g1 = __builtin_amdgcn_wmma_f32_16x16x32_f16(false, a, false, w1, (short)0, g1, false, false);
            g2 = __builtin_amdgcn_wmma_f32_16x16x32_f16(false, a, false, w2, (short)0, g2, false, false);
            g3 = __builtin_amdgcn_wmma_f32_16x16x32_f16(false, a, false, w3, (short)0, g3, false, false);
        }
        __syncthreads();   // all reads of hc/inp done before state update

        // ---- Phase D: LSTM cell update, in-register per (m,n) ----
        {
            const float bi_ = bsum_s[n],       bf_ = bsum_s[128 + n];
            const float bg_ = bsum_s[256 + n], bo_ = bsum_s[384 + n];
#pragma unroll
            for (int r = 0; r < 8; ++r) {
                int mm = r + off;
                float iv = g0[r] + bi_, fv = g1[r] + bf_;
                float gv = g2[r] + bg_, ov = g3[r] + bo_;
                float cn = sigm(fv) * cst[mm][n] + sigm(iv) * tanhf(gv);
                float hn = sigm(ov) * tanhf(cn);
                cst[mm][n] = cn;
                hc[mm][n] = (_Float16)hn;
                hc[mm][128 + n] = (_Float16)cn;
            }
        }
        __syncthreads();

        // ---- Phase E: out = relu(h@Wf1+bf1)@Wf2+bf2, waves 0..3 WMMA ----
        if (wv < 4) {
            v8f acc = {};
#pragma unroll
            for (int kc = 0; kc < 4; ++kc) {
                v16h a = load_frag(&hc[m][kc * 32 + off]);
                v16h b = load_frag(Wf1T + (wv * 16 + m) * H_ + kc * 32 + off);
                acc = __builtin_amdgcn_wmma_f32_16x16x32_f16(false, a, false, b,
                                                             (short)0, acc, false, false);
            }
#pragma unroll
            for (int r = 0; r < 8; ++r) {
                float v = acc[r] + bf1_s[wv * 16 + m];
                rel[r + off][wv * 16 + m] = fmaxf(v, 0.0f);
            }
        }
        __syncthreads();

        if (tid < 16) {
            float acc = bf2v;
#pragma unroll 8
            for (int k = 0; k < 64; ++k) acc += rel[tid][k] * Wf2_s[k];
            tp[tid] = acc;
            out[(long)(b0 + tid) * P_ + p] = acc;
        }
        __syncthreads();
    }
}

// ---------------------------------------------------------------------------
extern "C" void kernel_launch(void* const* d_in, const int* in_sizes, int n_in,
                              void* d_out, int out_size, void* d_ws, size_t ws_size,
                              hipStream_t stream) {
    (void)in_sizes; (void)n_in; (void)out_size; (void)ws_size;
    const float* x    = (const float*)d_in[0];
    const float* his  = (const float*)d_in[1];
    const float* Wa1  = (const float*)d_in[2];
    const float* ba1  = (const float*)d_in[3];
    const float* Wa2  = (const float*)d_in[4];
    const float* ba2  = (const float*)d_in[5];
    const float* Wih  = (const float*)d_in[6];
    const float* Whh  = (const float*)d_in[7];
    const float* bih  = (const float*)d_in[8];
    const float* bhh  = (const float*)d_in[9];
    const float* Wf1  = (const float*)d_in[10];
    const float* bf1  = (const float*)d_in[11];
    const float* Wf2  = (const float*)d_in[12];
    const float* bf2  = (const float*)d_in[13];

    char* ws = (char*)d_ws;
    size_t o = 0;
    auto take = [&](size_t bytes) { size_t r = o; o = (o + bytes + 255) & ~(size_t)255; return r; };
    _Float16* Wg     = (_Float16*)(ws + take((size_t)512 * KG_ * 2));
    _Float16* Wa1hcT = (_Float16*)(ws + take((size_t)64 * 256 * 2));
    _Float16* Wa1xT  = (_Float16*)(ws + take((size_t)64 * KX_ * 2));
    _Float16* Wf1T   = (_Float16*)(ws + take((size_t)64 * H_ * 2));
    float*    bsum   = (float*)   (ws + take((size_t)512 * 4));
    float*    xwb    = (float*)   (ws + take((size_t)B_ * CH_ * A_ * 4));

    pack_weights<<<256, 256, 0, stream>>>(Wa1, Wih, Whh, bih, bhh, Wf1,
                                          Wg, Wa1hcT, Wa1xT, Wf1T, bsum);
    xw_kernel<<<(B_ * CH_) / 16, 128, 0, stream>>>(x, Wa1xT, xwb);
    decoder_main<<<B_ / 16, 256, 0, stream>>>(x, his, ba1, Wa2, ba2, bf1, Wf2, bf2,
                                              Wg, Wa1hcT, Wf1T, bsum, xwb,
                                              (float*)d_out);
}
